// Attention_56530359550786
// MI455X (gfx1250) — compile-verified, matrix operand order checked by reading, not run on hardware
//
#include <hip/hip_runtime.h>

typedef _Float16 v16h __attribute__((ext_vector_type(16)));
typedef _Float16 v8h  __attribute__((ext_vector_type(8)));
typedef _Float16 v4h  __attribute__((ext_vector_type(4)));
typedef float    v8f  __attribute__((ext_vector_type(8)));
typedef float    v4f  __attribute__((ext_vector_type(4)));

#define B_      2
#define H_      16
#define L_      2048
#define D_      128
#define QTILES  (L_ / 64)

// softmax scale (1/sqrt(128)) folded together with log2(e) so we can use exp2
static constexpr float QSCALE = 0.08838834764831845f * 1.4426950408889634f;

#define KS_STRIDE 136   // halves; 272 B row pitch (16B aligned)
#define VT_STRIDE 72    // halves; 144 B row pitch (16B aligned)

// Raw hardware exp2: inputs here are always <= 0 (post max-subtraction), and
// v_exp_f32 flushes very negative inputs to 0 exactly as flash-attention needs,
// so we skip libm's ldexp/cndmask range fixup.
__device__ __forceinline__ float fast_exp2(float x) {
    return __builtin_amdgcn_exp2f(x);
}

// Load one 16x32 f16 A-fragment from LDS (two 16B-aligned ds_load_b128).
__device__ __forceinline__ v16h ld_frag(const _Float16* p) {
    const v8h lo = *(const v8h*)p;
    const v8h hg = *(const v8h*)(p + 16);
    return __builtin_shufflevector(lo, hg,
        0, 1, 2, 3, 4, 5, 6, 7, 8, 9, 10, 11, 12, 13, 14, 15);
}

__global__ __launch_bounds__(128)
void Attention_56530359550786_kernel(const float* __restrict__ Q,
                                     const float* __restrict__ K,
                                     const float* __restrict__ V,
                                     float* __restrict__ O) {
    __shared__ _Float16 ks[64 * KS_STRIDE];    // K tile, row-major [key][d]
    __shared__ _Float16 vt[128 * VT_STRIDE];   // V tile, transposed [d][key]

    const int bh   = blockIdx.x >> 5;   // b*H + h
    const int qi   = blockIdx.x & 31;   // q tile index (64 rows each)
    const int tid  = threadIdx.x;
    const int wave = tid >> 5;
    const int lane = tid & 31;
    const int ln   = lane & 15;
    const int hi   = lane >> 4;

    const size_t base = (size_t)bh * L_ * D_;
    const int q0   = qi * 64;
    const int qrow = q0 + wave * 16 + ln;   // this lane's query row (= its Sᵀ column)

    // ---- Load Q row as 4 B-fragments (d-chunks of 32) for Sᵀ = K·Qᵀ ----
    // f16 B-frag: lane l holds column n=l&15; element j<8 -> K=k0+8*hi+j, j>=8 -> K=k0+16+8*hi+(j-8)
    v16h qf[4];
    {
        const float* qp = Q + base + (size_t)qrow * D_;
        #pragma unroll
        for (int c = 0; c < 4; ++c) {
            const int d0 = 32 * c + 8 * hi;
            const v4f f0 = *(const v4f*)(qp + d0);
            const v4f f1 = *(const v4f*)(qp + d0 + 4);
            const v4f f2 = *(const v4f*)(qp + d0 + 16);
            const v4f f3 = *(const v4f*)(qp + d0 + 20);
            #pragma unroll
            for (int u = 0; u < 4; ++u) {
                qf[c][u]      = (_Float16)(f0[u] * QSCALE);
                qf[c][u + 4]  = (_Float16)(f1[u] * QSCALE);
                qf[c][u + 8]  = (_Float16)(f2[u] * QSCALE);
                qf[c][u + 12] = (_Float16)(f3[u] * QSCALE);
            }
        }
    }

    v8f o[8] = {};                 // Oᵀ accumulators: 8 d-tiles of 16x16 f32
    float m_run = -1e30f;          // per-lane (per-query) running max (log2 units)
    float l_run = 0.0f;            // per-lane running sum

    const v4f* kp4 = (const v4f*)(K + base);
    const v4f* vp4 = (const v4f*)(V + base);

    const _Float16* kbase = &ks[ln * KS_STRIDE + 8 * hi];
    const _Float16* vbase = &vt[ln * VT_STRIDE + 8 * hi];

    for (int j = 0; j <= qi; ++j) {
        const int j0 = j * 64;

        // ---- Cooperative tile load, two passes to cap in-flight VGPRs ----
        // Pass 1: K tile -> row-major f16 (16 b128 loads in flight = 64 VGPRs)
        #pragma unroll
        for (int it = 0; it < 16; ++it) {
            const int idx = tid + it * 128;
            const int row = idx >> 5;        // key 0..63
            const int c4  = idx & 31;        // float4 column 0..31
            const v4f kk = kp4[(size_t)(j0 + row) * 32 + c4];
            v4h kh;
            #pragma unroll
            for (int u = 0; u < 4; ++u) kh[u] = (_Float16)kk[u];
            *(v4h*)&ks[row * KS_STRIDE + c4 * 4] = kh;
        }
        // Pass 2: V tile -> transposed f16
        #pragma unroll
        for (int it = 0; it < 16; ++it) {
            const int idx = tid + it * 128;
            const int row = idx >> 5;
            const int c4  = idx & 31;
            const v4f vv = vp4[(size_t)(j0 + row) * 32 + c4];
            #pragma unroll
            for (int u = 0; u < 4; ++u)
                vt[(c4 * 4 + u) * VT_STRIDE + row] = (_Float16)vv[u];
        }
        // prefetch next KV tile into cache (L2-resident working set)
        if (j < qi) {
            const char* nk = (const char*)(kp4 + (size_t)(j0 + 64) * 32);
            const char* nv = (const char*)(vp4 + (size_t)(j0 + 64) * 32);
            __builtin_prefetch(nk + tid * 256, 0, 3);
            __builtin_prefetch(nv + tid * 256, 0, 3);
        }
        __syncthreads();

        // ---- Sᵀ = K_tile · Qᵀ : 16 WMMAs, A-fragment loads pipelined 2-deep ----
        v8f st[4] = {};
        {
            v16h a = ld_frag(kbase);
            #pragma unroll
            for (int f = 0; f < 16; ++f) {
                const int kt = f >> 2, c = f & 3;
                v16h an = a;
                if (f < 15) {
                    const int f1 = f + 1;
                    an = ld_frag(kbase + (f1 >> 2) * (16 * KS_STRIDE) + (f1 & 3) * 32);
                }
                st[kt] = __builtin_amdgcn_wmma_f32_16x16x32_f16(
                    false, a, false, qf[c], (short)0, st[kt], false, false);
                a = an;
            }
        }

        // ---- Causal mask (diagonal block only): key > query -> -inf ----
        if (j == qi) {
            #pragma unroll
            for (int kt = 0; kt < 4; ++kt) {
                #pragma unroll
                for (int r = 0; r < 8; ++r) {
                    const int key = j0 + kt * 16 + r + 8 * hi;
                    if (key > qrow) st[kt][r] = -1e30f;
                }
            }
        }

        // ---- Online softmax: one query per lane; keys split across hi-groups ----
        float mb = -1e30f;
        #pragma unroll
        for (int kt = 0; kt < 4; ++kt)
            #pragma unroll
            for (int r = 0; r < 8; ++r) mb = fmaxf(mb, st[kt][r]);
        mb = fmaxf(mb, __shfl_xor(mb, 16, 32));
        const float m_new = fmaxf(m_run, mb);
        const float alpha = fast_exp2(m_run - m_new);
        float ls = 0.0f;
        #pragma unroll
        for (int kt = 0; kt < 4; ++kt)
            #pragma unroll
            for (int r = 0; r < 8; ++r) {
                const float p = fast_exp2(st[kt][r] - m_new);
                st[kt][r] = p;
                ls += p;
            }
        ls += __shfl_xor(ls, 16, 32);
        l_run = l_run * alpha + ls;
        m_run = m_new;
        #pragma unroll
        for (int dt = 0; dt < 8; ++dt)
            #pragma unroll
            for (int r = 0; r < 8; ++r) o[dt][r] *= alpha;

        // ---- Pack Pᵀ B-fragments: C-layout of Sᵀ maps 1:1 onto f16 B-layout ----
        v16h pf[2];
        #pragma unroll
        for (int kc = 0; kc < 2; ++kc)
            #pragma unroll
            for (int jx = 0; jx < 8; ++jx) {
                pf[kc][jx]     = (_Float16)st[2 * kc][jx];
                pf[kc][jx + 8] = (_Float16)st[2 * kc + 1][jx];
            }

        // ---- Oᵀ += Vᵀ · Pᵀ : 16 WMMAs, A-fragment loads pipelined 2-deep ----
        {
            v16h a = ld_frag(vbase);
            #pragma unroll
            for (int f = 0; f < 16; ++f) {
                const int dt = f >> 1, kc = f & 1;
                v16h an = a;
                if (f < 15) {
                    const int f1 = f + 1;
                    an = ld_frag(vbase + (f1 >> 1) * (16 * VT_STRIDE) + (f1 & 1) * 32);
                }
                o[dt] = __builtin_amdgcn_wmma_f32_16x16x32_f16(
                    false, a, false, pf[kc], (short)0, o[dt], false, false);
                a = an;
            }
        }
        __syncthreads();
    }

    // ---- Normalize and write O: lane's regs r=0..7 are 8 consecutive d values ----
    const float invl = 1.0f / l_run;
    float* op = O + base + (size_t)qrow * D_;
    #pragma unroll
    for (int dt = 0; dt < 8; ++dt) {
        const int d0 = dt * 16 + 8 * hi;
        v4f a, b;
        #pragma unroll
        for (int u = 0; u < 4; ++u) {
            a[u] = o[dt][u] * invl;
            b[u] = o[dt][u + 4] * invl;
        }
        *(v4f*)(op + d0)     = a;
        *(v4f*)(op + d0 + 4) = b;
    }
}

extern "C" void kernel_launch(void* const* d_in, const int* in_sizes, int n_in,
                              void* d_out, int out_size, void* d_ws, size_t ws_size,
                              hipStream_t stream) {
    const float* q = (const float*)d_in[0];
    const float* k = (const float*)d_in[1];
    const float* v = (const float*)d_in[2];
    float* out = (float*)d_out;
    dim3 grid(B_ * H_ * QTILES);   // 2*16*32 = 1024 blocks
    dim3 block(128);               // 4 waves x wave32; 16 q-rows per wave
    Attention_56530359550786_kernel<<<grid, block, 0, stream>>>(q, k, v, out);
}
